// RelativeBias_9294309228545
// MI455X (gfx1250) — matrix operand order repaired and checked by reading, not compiled, not run on hardware
//
#include <hip/hip_runtime.h>
#include <hip/hip_bf16.h>

typedef __attribute__((ext_vector_type(16))) _Float16 v16h;
typedef __attribute__((ext_vector_type(8)))  float    v8f;
typedef __attribute__((ext_vector_type(4)))  float    v4f;

#define NPTS  1024
#define HEADS 16
// freqs = linspace(0,1,32)/0.1 -> freq[k] = k / 3.1
#define FREQ_SCALE (1.0f / 3.1f)

__global__ __launch_bounds__(256) void relbias_wmma_kernel(
    const float* __restrict__ vec_map,   // (4, 1024, 3)
    const float* __restrict__ kern,      // (64, 16)
    const float* __restrict__ bias,      // (16,)
    float* __restrict__ out,             // (4, 1024, 1024, 16)
    int num_tiles)
{
  // one 16x16 f32 tile (1 KB) per wave, 8 waves per block
  __shared__ __align__(16) float lds[8 * 256];

  const int lane   = threadIdx.x & 31;
  const int wid    = threadIdx.x >> 5;
  const int gwave  = blockIdx.x * 8 + wid;
  const int nwaves = gridDim.x * 8;

  const int col = lane & 15;        // head index (B/C/D column), and A row M
  const int hi  = (lane >> 4) & 1;  // lane half

  // ---- Preload B matrices (dense kernel) in contiguous-K order (k = 16*half + e) ----
  // lane holds column N=col; element e -> K = kB + e
  const int kB = hi ? 16 : 0;
  v16h b_sin, b_cos;
#pragma unroll
  for (int e = 0; e < 16; ++e) {
    b_sin[e] = (_Float16)kern[(kB + e) * HEADS + col];        // kernel rows 0..31 (sin)
    b_cos[e] = (_Float16)kern[(32 + kB + e) * HEADS + col];   // kernel rows 32..63 (cos)
  }
  const float bias_v = bias[col];

  // A-matrix 16x32 f16, same contiguous K order as B: lane half -> K base 0 or 16;
  // element e -> K = kbA + e
  const int kbA = hi ? 16 : 0;

  float* tl = lds + wid * 256;

  for (int tile = gwave; tile < num_tiles; tile += nwaves) {
    const int row0   = tile << 4;          // flat (b*N + n)*N + m, tile-aligned
    const int bn     = row0 >> 10;         // b*1024 + n  (16 | 1024, so n fixed per tile)
    const int m_base = row0 & (NPTS - 1);

    // per-lane geometry: this lane's A row is m = m_base + col
    const float* vn = vec_map + (size_t)bn * 3;
    const float* vm = vec_map + ((size_t)(bn & ~(NPTS - 1)) + m_base + col) * 3;
    float dot = vn[0] * vm[0] + vn[1] * vm[1] + vn[2] * vm[2];
    dot = fminf(fmaxf(dot, 0.0f), 1.0f);
    const float angle = acosf(dot) * 1000.0f;   // / 0.001

    // Build A_sin / A_cos tiles; each lane computes 16 distinct sincos pairs,
    // lanes L and L+16 cover disjoint K subsets -> no redundant transcendentals.
    v16h a_sin, a_cos;
#pragma unroll
    for (int e = 0; e < 16; ++e) {
      const int   k   = kbA + e;
      const float arg = angle * ((float)k * FREQ_SCALE);
      float s, c;
      __sincosf(arg, &s, &c);
      a_sin[e] = (_Float16)s;
      a_cos[e] = (_Float16)c;
    }

    // D = A_sin * B_sin + A_cos * B_cos  (16x16 f32 accum)
    v8f acc = {};
    acc = __builtin_amdgcn_wmma_f32_16x16x32_f16(
        /*neg_a=*/false, a_sin, /*neg_b=*/false, b_sin,
        /*c_mod=*/(short)0, acc, /*reuse_a=*/false, /*reuse_b=*/false);
    acc = __builtin_amdgcn_wmma_f32_16x16x32_f16(
        false, a_cos, false, b_cos, (short)0, acc, false, false);

    // Stage tile through LDS to turn the 64B-strided D layout into
    // fully contiguous 1KB global stores (2 x b128 per lane).
#pragma unroll
    for (int i = 0; i < 8; ++i) {
      const int M = i + hi * 8;                 // C/D row for VGPR i
      tl[M * 16 + col] = acc[i] + bias_v;
    }
    // intra-wave LDS visibility (wave32 lockstep; no block barrier needed)
    asm volatile("s_wait_dscnt 0" ::: "memory");

    const v4f* lp = (const v4f*)tl;
    v4f r0 = lp[lane * 2 + 0];
    v4f r1 = lp[lane * 2 + 1];
    // Output is a 256MB write-once stream: use non-temporal stores so it
    // doesn't rinse through / evict the 192MB L2 (TH_STORE_NT).
    v4f* op = (v4f*)(out + (size_t)row0 * HEADS);
    __builtin_nontemporal_store(r0, op + lane * 2 + 0);
    __builtin_nontemporal_store(r1, op + lane * 2 + 1);
    // compiler's dscnt tracking orders the ds_loads before register use, and
    // same-wave LDS ops are in-order, so next iteration's stores are safe.
  }
}

extern "C" void kernel_launch(void* const* d_in, const int* in_sizes, int n_in,
                              void* d_out, int out_size, void* d_ws, size_t ws_size,
                              hipStream_t stream) {
  (void)in_sizes; (void)n_in; (void)d_ws; (void)ws_size;
  const float* vec_map = (const float*)d_in[0];
  const float* kern    = (const float*)d_in[1];
  const float* bias    = (const float*)d_in[2];
  float* out = (float*)d_out;

  const int num_tiles = out_size / (16 * HEADS);   // 262,144 tiles of 16 rows x 16 heads
  const int blocks = 2048;                         // 16,384 waves -> 16 tiles/wave
  relbias_wmma_kernel<<<blocks, 256, 0, stream>>>(vec_map, kern, bias, out, num_tiles);
}
